// MultiHeadAttention_57878979281572
// MI455X (gfx1250) — matrix operand order repaired and checked by reading, not compiled, not run on hardware
//
#include <hip/hip_runtime.h>
#include <hip/hip_bf16.h>
#include <stdint.h>

// ---------------------------------------------------------------------------
// MI455X (gfx1250) fused multi-head attention, wave32 + WMMA f16->f32,
// async global->LDS staging (ASYNCcnt), double-buffered GEMM tiles.
// B=2, S=2048, D=1024, H=16, Dh=64.
// ---------------------------------------------------------------------------

#define Bq 2
#define Sq 2048
#define Dm 1024
#define Hh 16
#define Dh 64
#define NEGMASK (-9.0e10f)

typedef __attribute__((ext_vector_type(16))) _Float16 v16h;
typedef __attribute__((ext_vector_type(8)))  _Float16 v8h;
typedef __attribute__((ext_vector_type(8)))  float    v8f;

union AFrag { v16h v; v8h h[2]; };

__device__ inline v8f wmma_f16(v16h a, v16h b, v8f c) {
  // D = A(16x32 f16) x B(32x16 f16) + C(16x16 f32)
  return __builtin_amdgcn_wmma_f32_16x16x32_f16(false, a, false, b,
                                                (short)0, c, false, false);
}

// Async global->LDS copy of 16 bytes per lane (GLOBAL_LOAD_ASYNC_TO_LDS_B128,
// tracked by ASYNCcnt). The LDS operand is the wave-relative byte offset
// (low 32 bits of the generic pointer to a __shared__ object).
__device__ inline void async_b128(const _Float16* g, const _Float16* l) {
  uint32_t loff = (uint32_t)(uintptr_t)l;
  asm volatile("global_load_async_to_lds_b128 %0, %1, off"
               :: "v"(loff), "v"(g) : "memory");
}
__device__ inline void wait_async0() {
  asm volatile("s_wait_asynccnt 0x0" ::: "memory");
}

// ---------------------------------------------------------------------------
// Pass 0a: elementwise f32 -> f16 conversion (Q,K,V), b128 in / b128 out.
// ---------------------------------------------------------------------------
__global__ __launch_bounds__(256) void cvt_f32_f16(
    const float* __restrict__ src, _Float16* __restrict__ dst) {
  const size_t i = ((size_t)blockIdx.x * 256 + threadIdx.x) * 8;
  float4 f0 = ((const float4*)(src + i))[0];
  float4 f1 = ((const float4*)(src + i))[1];
  v8h h = {(_Float16)f0.x, (_Float16)f0.y, (_Float16)f0.z, (_Float16)f0.w,
           (_Float16)f1.x, (_Float16)f1.y, (_Float16)f1.z, (_Float16)f1.w};
  *(v8h*)(dst + i) = h;
}

// ---------------------------------------------------------------------------
// Pass 0b: Wt[k][n] = (f16) W[n][k]  (1024x1024), LDS-tiled transpose.
// ---------------------------------------------------------------------------
__global__ __launch_bounds__(256) void transpose_w_f16(
    const float* __restrict__ W, _Float16* __restrict__ Wt) {
  __shared__ float tile[32][33];
  const int tx = threadIdx.x;        // 0..31
  const int ty = threadIdx.y;        // 0..7
  const int bk = blockIdx.x * 32;    // k tile
  const int bn = blockIdx.y * 32;    // n tile
#pragma unroll
  for (int i = 0; i < 4; ++i) {
    int n = bn + ty + 8 * i;
    tile[ty + 8 * i][tx] = W[(size_t)n * Dm + bk + tx];  // coalesced in k
  }
  __syncthreads();
#pragma unroll
  for (int i = 0; i < 4; ++i) {
    int k = bk + ty + 8 * i;
    Wt[(size_t)k * Dm + bn + tx] = (_Float16)tile[tx][ty + 8 * i];
  }
}

// ---------------------------------------------------------------------------
// Pass 1/3: out[m,n] = sum_k A[m,k] * Wt[k,n] + bias[n]
//   A: f16 (M x 1024) row-major, Wt: f16 (1024 x 1024) k-major.
//   Block: 256 threads = 8 waves; tile 128(M) x 64(N); K step 32.
//   Both operand tiles staged via async global->LDS b128 copies, double
//   buffered so the copy of tile kt+1 overlaps the WMMAs of tile kt.
//   K-loop forced to unroll(1) with byte-offset buffer select so the
//   accumulators stay pinned in one VGPR range (no phi-copy dual_movs).
//   mode 0: store f16 q-layout  [b][h][s][dh]
//   mode 1: store f16 kT-layout [b][h][dh][s]
//   mode 2: store f16 v-layout  [b][h][s][dh]
//   mode 3: store f32 [m][n] (final output)
// ---------------------------------------------------------------------------
#define APAD 40  // halves per A row in LDS (bank-conflict-free 16B reads)
#define BPAD 72  // halves per B row in LDS
#define ABUF (128 * APAD)
#define BBUF (32 * BPAD)

__global__ __launch_bounds__(256) void gemm_xwT(
    const _Float16* __restrict__ A, const _Float16* __restrict__ Wt,
    const float* __restrict__ bias, void* __restrict__ out, int mode) {
  __shared__ __align__(16) _Float16 lds_a[2 * ABUF];  // 2 x 10240 B
  __shared__ __align__(16) _Float16 lds_b[2 * BBUF];  // 2 x  4608 B

  const int tid  = threadIdx.x;
  const int lane = tid & 31;
  const int wid  = tid >> 5;          // 0..7
  const int hi   = lane >> 4;         // 0/1
  const int lo   = lane & 15;
  const int m0   = blockIdx.y * 128;
  const int n0   = blockIdx.x * 64;

  // staging coordinates (per thread: 32B of A, 16B of B per tile)
  const int arow = tid >> 1, akof = (tid & 1) * 16;
  const int brow = tid >> 3, bcol = (tid & 7) * 8;

  auto stage = [&](int kt, int buf) {
    const int k0 = kt * 32;
    const _Float16* ga = A + (size_t)(m0 + arow) * Dm + k0 + akof;
    const _Float16* la = lds_a + buf * ABUF + arow * APAD + akof;
    async_b128(ga, la);
    async_b128(ga + 8, la + 8);
    const _Float16* gb = Wt + (size_t)(k0 + brow) * Dm + n0 + bcol;
    async_b128(gb, lds_b + buf * BBUF + brow * BPAD + bcol);
  };

  const v8f vzero = {0.f, 0.f, 0.f, 0.f, 0.f, 0.f, 0.f, 0.f};
  v8f acc0 = vzero, acc1 = vzero, acc2 = vzero, acc3 = vzero;

  const int NT = Dm / 32;
  stage(0, 0);
#pragma unroll 1
  for (int kt = 0; kt < NT; ++kt) {
    const int bi = kt & 1;
    wait_async0();         // copies for buffer bi (this wave) complete
    __syncthreads();       // all waves: copies done AND prior compute done
    if (kt + 1 < NT) stage(kt + 1, bi ^ 1);  // overlap with WMMAs below

    // A fragment (16x32): lane lo = row, hi selects K sub-chunks
    AFrag af;
    {
      const _Float16* ab = lds_a + bi * ABUF + (wid * 16 + lo) * APAD;
      af.h[0] = *(const v8h*)(ab + hi * 8);
      af.h[1] = *(const v8h*)(ab + 16 + hi * 8);
    }
    const _Float16* bbp = lds_b + bi * BBUF + lo;
    AFrag b0, b1, b2, b3;
#pragma unroll
    for (int s = 0; s < 16; ++s) {
      const int kr = (s & 7) + 8 * hi + 16 * (s >> 3);
      b0.v[s] = bbp[kr * BPAD +  0];
      b1.v[s] = bbp[kr * BPAD + 16];
      b2.v[s] = bbp[kr * BPAD + 32];
      b3.v[s] = bbp[kr * BPAD + 48];
    }
    acc0 = wmma_f16(af.v, b0.v, acc0);
    acc1 = wmma_f16(af.v, b1.v, acc1);
    acc2 = wmma_f16(af.v, b2.v, acc2);
    acc3 = wmma_f16(af.v, b3.v, acc3);
  }

  // epilogue
  const int mrow0 = m0 + wid * 16;
  v8f acc[4] = {acc0, acc1, acc2, acc3};
#pragma unroll
  for (int j = 0; j < 4; ++j) {
    const int n = n0 + j * 16 + lo;
    const float bn = bias[n];
    const int h  = n >> 6;
    const int dh = n & 63;
#pragma unroll
    for (int r = 0; r < 8; ++r) {
      const int m = mrow0 + r + 8 * hi;
      const float val = acc[j][r] + bn;
      const int bb = m >> 11;       // m = b*S + s
      const int ss = m & (Sq - 1);
      if (mode == 3) {
        ((float*)out)[(size_t)m * Dm + n] = val;
      } else if (mode == 1) {
        ((_Float16*)out)[(((size_t)(bb * Hh + h)) * Dh + dh) * Sq + ss] =
            (_Float16)val;
      } else {
        ((_Float16*)out)[(((size_t)(bb * Hh + h)) * Sq + ss) * Dh + dh] =
            (_Float16)val;
      }
    }
  }
}

// ---------------------------------------------------------------------------
// Pass 2: flash attention. Block = 128 threads = 4 waves; each wave owns 16
// query rows; online softmax over 32-key blocks; ctx (f16) out in (b,s,d).
// ---------------------------------------------------------------------------
__global__ __launch_bounds__(128) void attn_flash(
    const _Float16* __restrict__ qh, const _Float16* __restrict__ kT,
    const _Float16* __restrict__ vh, const uint8_t* __restrict__ mask,
    _Float16* __restrict__ ctx) {
  __shared__ __align__(16) _Float16 lds_p[4][16 * APAD];

  const int lane = threadIdx.x & 31;
  const int wid  = threadIdx.x >> 5;  // 0..3
  const int hi   = lane >> 4;
  const int lo   = lane & 15;
  const int b    = blockIdx.z;
  const int h    = blockIdx.y;
  const int q0   = blockIdx.x * 64 + wid * 16;

  const _Float16* qbase = qh + ((size_t)(b * Hh + h)) * Sq * Dh;
  const _Float16* kbase = kT + ((size_t)(b * Hh + h)) * Dh * Sq;
  const _Float16* vbase = vh + ((size_t)(b * Hh + h)) * Sq * Dh;
  const uint8_t*  mbase = mask + (size_t)b * Sq * Sq;

  // Q fragments: two 16x32 tiles covering dh=0..63
  AFrag aq[2];
#pragma unroll
  for (int c = 0; c < 2; ++c) {
    const _Float16* qp = qbase + (size_t)(q0 + lo) * Dh + 32 * c;
    aq[c].h[0] = *(const v8h*)(qp + hi * 8);
    aq[c].h[1] = *(const v8h*)(qp + 16 + hi * 8);
  }

  const v8f vzero = {0.f, 0.f, 0.f, 0.f, 0.f, 0.f, 0.f, 0.f};
  v8f o[4] = {vzero, vzero, vzero, vzero};
  float mrow[8], lrow[8];
#pragma unroll
  for (int r = 0; r < 8; ++r) { mrow[r] = -3.0e38f; lrow[r] = 0.f; }

#pragma unroll 1
  for (int kb = 0; kb < Sq / 32; ++kb) {
    const int key0 = kb * 32;
    v8f s[2] = {vzero, vzero};
    // scores: 16q x 32k, contraction over dh=64 (2 WMMAs per key sub-tile)
#pragma unroll
    for (int j = 0; j < 2; ++j) {
#pragma unroll
      for (int c = 0; c < 2; ++c) {
        const _Float16* kp = kbase + (size_t)(32 * c) * Sq + key0 + 16 * j + lo;
        AFrag bf;
#pragma unroll
        for (int s = 0; s < 16; ++s)
          bf.v[s] = kp[(size_t)((s & 7) + 8 * hi + 16 * (s >> 3)) * Sq];
        s[j] = wmma_f16(aq[c].v, bf.v, s[j]);
      }
    }
    // scale + additive mask, block row-max
    float bm[8];
#pragma unroll
    for (int j = 0; j < 2; ++j) {
#pragma unroll
      for (int r = 0; r < 8; ++r) {
        const int qrow = q0 + r + 8 * hi;
        const int kcol = key0 + 16 * j + lo;
        float val = s[j][r] * 0.125f;  // 1/sqrt(64)
        if (mbase[(size_t)qrow * Sq + kcol]) val += NEGMASK;
        s[j][r] = val;
        bm[r] = (j == 0) ? val : fmaxf(bm[r], val);
      }
    }
#pragma unroll
    for (int r = 0; r < 8; ++r) {
      bm[r] = fmaxf(bm[r], __shfl_xor(bm[r], 1, 32));
      bm[r] = fmaxf(bm[r], __shfl_xor(bm[r], 2, 32));
      bm[r] = fmaxf(bm[r], __shfl_xor(bm[r], 4, 32));
      bm[r] = fmaxf(bm[r], __shfl_xor(bm[r], 8, 32));
    }
    // online softmax update; write P (f16) to this wave's LDS slice
    float corr[8], rs[8];
#pragma unroll
    for (int r = 0; r < 8; ++r) {
      const float mn = fmaxf(mrow[r], bm[r]);
      corr[r] = __expf(mrow[r] - mn);
      mrow[r] = mn;
      rs[r] = 0.f;
    }
#pragma unroll
    for (int j = 0; j < 2; ++j) {
#pragma unroll
      for (int r = 0; r < 8; ++r) {
        const float p = __expf(s[j][r] - mrow[r]);
        rs[r] += p;
        lds_p[wid][(r + 8 * hi) * APAD + 16 * j + lo] = (_Float16)p;
      }
    }
#pragma unroll
    for (int r = 0; r < 8; ++r) {
      rs[r] += __shfl_xor(rs[r], 1, 32);
      rs[r] += __shfl_xor(rs[r], 2, 32);
      rs[r] += __shfl_xor(rs[r], 4, 32);
      rs[r] += __shfl_xor(rs[r], 8, 32);
      lrow[r] = lrow[r] * corr[r] + rs[r];
    }
    // rescale running context (C-layout rows match corr[] rows)
#pragma unroll
    for (int j = 0; j < 4; ++j)
#pragma unroll
      for (int r = 0; r < 8; ++r) o[j][r] *= corr[r];

    // P as A-fragment (16x32) from LDS (wave-private slice, in-order DS)
    AFrag pa;
    {
      const _Float16* pp = &lds_p[wid][lo * APAD];
      pa.h[0] = *(const v8h*)(pp + hi * 8);
      pa.h[1] = *(const v8h*)(pp + 16 + hi * 8);
    }
    // ctx += P(16x32) x V(32x64)
#pragma unroll
    for (int j = 0; j < 4; ++j) {
      const _Float16* vp = vbase + (size_t)key0 * Dh + 16 * j + lo;
      AFrag bf;
#pragma unroll
      for (int s = 0; s < 16; ++s)
        bf.v[s] = vp[(size_t)((s & 7) + 8 * hi + 16 * (s >> 3)) * Dh];
      o[j] = wmma_f16(pa.v, bf.v, o[j]);
    }
  }

  // normalize and store ctx (f16) in (b, s, h*64+dh)
#pragma unroll
  for (int j = 0; j < 4; ++j) {
#pragma unroll
    for (int r = 0; r < 8; ++r) {
      const int srow = q0 + r + 8 * hi;
      const int dcol = h * Dh + 16 * j + lo;
      ctx[((size_t)b * Sq + srow) * Dm + dcol] = (_Float16)(o[j][r] / lrow[r]);
    }
  }
}

// ---------------------------------------------------------------------------
// Host launcher. Inputs: Q,K,V,mask,Wq,bq,Wk,bk,Wv,bv,Wo,bo
// ---------------------------------------------------------------------------
extern "C" void kernel_launch(void* const* d_in, const int* in_sizes, int n_in,
                              void* d_out, int out_size, void* d_ws,
                              size_t ws_size, hipStream_t stream) {
  const float* Q  = (const float*)d_in[0];
  const float* K  = (const float*)d_in[1];
  const float* V  = (const float*)d_in[2];
  const uint8_t* mask = (const uint8_t*)d_in[3];
  const float* Wq = (const float*)d_in[4];
  const float* bq = (const float*)d_in[5];
  const float* Wk = (const float*)d_in[6];
  const float* bk = (const float*)d_in[7];
  const float* Wv = (const float*)d_in[8];
  const float* bv = (const float*)d_in[9];
  const float* Wo = (const float*)d_in[10];
  const float* bo = (const float*)d_in[11];

  char* ws = (char*)d_ws;
  const size_t MB = 1024u * 1024u;
  _Float16* Wt_q = (_Float16*)(ws + 0 * MB);   // 2MB each
  _Float16* Wt_k = (_Float16*)(ws + 2 * MB);
  _Float16* Wt_v = (_Float16*)(ws + 4 * MB);
  _Float16* Wt_o = (_Float16*)(ws + 6 * MB);
  _Float16* ctxh = (_Float16*)(ws + 8 * MB);   // 8MB
  _Float16* Qh16 = (_Float16*)(ws + 16 * MB);  // 8MB each
  _Float16* Kh16 = (_Float16*)(ws + 24 * MB);
  _Float16* Vh16 = (_Float16*)(ws + 32 * MB);
  _Float16* qh   = (_Float16*)(ws + 40 * MB);  // 8MB each
  _Float16* kT   = (_Float16*)(ws + 48 * MB);
  _Float16* vh   = (_Float16*)(ws + 56 * MB);

  const size_t nQKV = (size_t)Bq * Sq * Dm;  // 4M elements

  // Pass 0: f32->f16 activations + transposed f16 weights
  {
    dim3 g((unsigned)(nQKV / (256 * 8))), blk(256);
    hipLaunchKernelGGL(cvt_f32_f16, g, blk, 0, stream, Q, Qh16);
    hipLaunchKernelGGL(cvt_f32_f16, g, blk, 0, stream, K, Kh16);
    hipLaunchKernelGGL(cvt_f32_f16, g, blk, 0, stream, V, Vh16);
    dim3 gt(Dm / 32, Dm / 32), bt(32, 8);
    hipLaunchKernelGGL(transpose_w_f16, gt, bt, 0, stream, Wq, Wt_q);
    hipLaunchKernelGGL(transpose_w_f16, gt, bt, 0, stream, Wk, Wt_k);
    hipLaunchKernelGGL(transpose_w_f16, gt, bt, 0, stream, Wv, Wt_v);
    hipLaunchKernelGGL(transpose_w_f16, gt, bt, 0, stream, Wo, Wt_o);
  }
  // Pass 1: QKV projections (WMMA GEMM, async-staged tiles)
  {
    dim3 g(Dm / 64, (Bq * Sq) / 128), blk(256);
    hipLaunchKernelGGL(gemm_xwT, g, blk, 0, stream, Qh16, Wt_q, bq, (void*)qh, 0);
    hipLaunchKernelGGL(gemm_xwT, g, blk, 0, stream, Kh16, Wt_k, bk, (void*)kT, 1);
    hipLaunchKernelGGL(gemm_xwT, g, blk, 0, stream, Vh16, Wt_v, bv, (void*)vh, 2);
  }
  // Pass 2: flash attention
  {
    dim3 g(Sq / 64, Hh, Bq), blk(128);
    hipLaunchKernelGGL(attn_flash, g, blk, 0, stream, qh, kT, vh, mask, ctxh);
  }
  // Pass 3: output projection -> f32 d_out
  {
    dim3 g(Dm / 64, (Bq * Sq) / 128), blk(256);
    hipLaunchKernelGGL(gemm_xwT, g, blk, 0, stream, ctxh, Wt_o, bo, d_out, 3);
  }
}
